// HeteroGATEncoderLinear_15805479649918
// MI455X (gfx1250) — compile-verified, matrix operand order checked by reading, not run on hardware
//
#include <hip/hip_runtime.h>

// ---------------------------------------------------------------------------
// Problem constants (from the reference)
// ---------------------------------------------------------------------------
#define NN   30000
#define DD   256
#define EE   300000
#define TT   2
#define HID_ 64
#define OUT_ 256
#define HEADS_ 8
#define F1_  (HEADS_*HID_)        // 512
#define EP_  (EE + NN)            // edges + self loops = 330000

typedef __attribute__((ext_vector_type(16))) __bf16 v16bf;
typedef __attribute__((ext_vector_type(8)))  float  v8f;

union Frag16 {
    v16bf v;
    uint4 q[2];
    unsigned short s[16];
};

__device__ __forceinline__ unsigned short f2bf(float f) {
    unsigned u = __float_as_uint(f);
    unsigned r = u + 0x7fffu + ((u >> 16) & 1u);   // round to nearest even
    return (unsigned short)(r >> 16);
}

// monotone float <-> ordered-uint encoding for atomicMax on floats
__device__ __forceinline__ unsigned encf(float f) {
    int i = __float_as_int(f);
    return (i >= 0) ? ((unsigned)i | 0x80000000u) : (unsigned)(~i);
}
__device__ __forceinline__ float decf(unsigned u) {
    int i = (u & 0x80000000u) ? (int)(u & 0x7fffffffu) : ~(int)u;
    return __int_as_float(i);
}

// ---------------------------------------------------------------------------
// Elementwise helpers
// ---------------------------------------------------------------------------
__global__ void fill_u32(unsigned* __restrict__ p, unsigned v, int n) {
    int i = blockIdx.x * blockDim.x + threadIdx.x;
    if (i < n) p[i] = v;
}

__global__ void convert_bf16(const float* __restrict__ in,
                             unsigned short* __restrict__ out, int n) {
    int i = blockIdx.x * blockDim.x + threadIdx.x;
    if (i < n) out[i] = f2bf(in[i]);
}

// in: K x Nc row-major f32 ; out: Nc x K row-major bf16 (i.e. B transposed)
__global__ void convert_transpose_bf16(const float* __restrict__ in,
                                       unsigned short* __restrict__ out,
                                       int K, int Nc) {
    int i = blockIdx.x * blockDim.x + threadIdx.x;
    if (i < K * Nc) {
        int k = i / Nc, n = i - k * Nc;
        out[(size_t)n * K + k] = f2bf(in[i]);
    }
}

// h16[i] = bf16( elu( h[i] + b0[col] + b1[col] ) )
__global__ void elu_bias_convert(const float* __restrict__ h,
                                 const float* __restrict__ b0,
                                 const float* __restrict__ b1,
                                 unsigned short* __restrict__ h16,
                                 int n, int F) {
    int i = blockIdx.x * blockDim.x + threadIdx.x;
    if (i >= n) return;
    int c = i % F;
    float v = h[i] + b0[c] + b1[c];
    v = (v > 0.f) ? v : (__expf(v) - 1.f);
    h16[i] = f2bf(v);
}

// ---------------------------------------------------------------------------
// bf16 WMMA GEMM:  C[M x Nc] = A[M x K] * B[K x Nc] + bias[Nc]
//   A  : bf16 row-major
//   Bt : bf16, B transposed (Nc x K row-major)
// Block: 256 threads (8 wave32), block tile 128(M) x 64(N).
// Waves: 4 M-strips x 2 N-halves; each wave computes a 32x32 tile with
// 2x2 v_wmma_f32_16x16x32_bf16 accumulators (each A/B fragment reused twice).
// K step = 32.
// ---------------------------------------------------------------------------
__global__ __launch_bounds__(256)
void wmma_gemm_bias(const unsigned short* __restrict__ A,
                    const unsigned short* __restrict__ Bt,
                    const float* __restrict__ bias,
                    float* __restrict__ C,
                    int M, int Nc, int K) {
    const int wave = threadIdx.x >> 5;
    const int lane = threadIdx.x & 31;
    const int lm = lane & 15;       // row/col within 16
    const int lh = lane >> 4;       // lane half (K phase)
    const int ms = wave & 3;        // M strip (4 strips of 32)
    const int nh = wave >> 2;       // N half  (2 halves of 32)

    const int mbase = blockIdx.x * 128 + ms * 32;
    const int n0 = blockIdx.y * 64 + nh * 32;
    const int n1 = n0 + 16;

    int arow0 = mbase + lm;      if (arow0 >= M) arow0 = M - 1;  // clamp; stores guarded
    int arow1 = mbase + 16 + lm; if (arow1 >= M) arow1 = M - 1;

    const unsigned short* ap0 = A  + (size_t)arow0     * K + lh * 8;
    const unsigned short* ap1 = A  + (size_t)arow1     * K + lh * 8;
    const unsigned short* bp0 = Bt + (size_t)(n0 + lm) * K + lh * 8;
    const unsigned short* bp1 = Bt + (size_t)(n1 + lm) * K + lh * 8;

    const float bv0 = bias ? bias[n0 + lm] : 0.f;
    const float bv1 = bias ? bias[n1 + lm] : 0.f;
    v8f c00, c01, c10, c11;
#pragma unroll
    for (int r = 0; r < 8; ++r) { c00[r] = bv0; c01[r] = bv1; c10[r] = bv0; c11[r] = bv1; }

    for (int k0 = 0; k0 < K; k0 += 32) {
        Frag16 a0, a1, b0, b1;
        // 16-bit A layout (ISA 7.12.2): lane half lh -> K = k0+8*lh..+7 and +16
        a0.q[0] = *(const uint4*)(ap0 + k0);
        a0.q[1] = *(const uint4*)(ap0 + k0 + 16);
        a1.q[0] = *(const uint4*)(ap1 + k0);
        a1.q[1] = *(const uint4*)(ap1 + k0 + 16);
        b0.q[0] = *(const uint4*)(bp0 + k0);
        b0.q[1] = *(const uint4*)(bp0 + k0 + 16);
        b1.q[0] = *(const uint4*)(bp1 + k0);
        b1.q[1] = *(const uint4*)(bp1 + k0 + 16);
        c00 = __builtin_amdgcn_wmma_f32_16x16x32_bf16(false, a0.v, false, b0.v,
                                                      (short)0, c00, false, false);
        c01 = __builtin_amdgcn_wmma_f32_16x16x32_bf16(false, a0.v, false, b1.v,
                                                      (short)0, c01, false, false);
        c10 = __builtin_amdgcn_wmma_f32_16x16x32_bf16(false, a1.v, false, b0.v,
                                                      (short)0, c10, false, false);
        c11 = __builtin_amdgcn_wmma_f32_16x16x32_bf16(false, a1.v, false, b1.v,
                                                      (short)0, c11, false, false);
    }

    // C/D layout: lanes 0-15 col=lane rows 0..7 ; lanes 16-31 col=lane-16 rows 8..15
    const int colA = n0 + lm, colB = n1 + lm;
#pragma unroll
    for (int r = 0; r < 8; ++r) {
        int g0 = mbase + lh * 8 + r;
        int g1 = g0 + 16;
        if (g0 < M) {
            C[(size_t)g0 * Nc + colA] = c00[r];
            C[(size_t)g0 * Nc + colB] = c01[r];
        }
        if (g1 < M) {
            C[(size_t)g1 * Nc + colA] = c10[r];
            C[(size_t)g1 * Nc + colB] = c11[r];
        }
    }
}

// ---------------------------------------------------------------------------
// GATv2 attention stages. Edges e in [0,EP): e<E -> ei[e]/ei[E+e], else self.
// ---------------------------------------------------------------------------

// One wave32 per (edge, head): lanes stride channels (coalesced 128B loads),
// then 5-step shfl_xor reduction across the wave.
__global__ void gat_scores(const float* __restrict__ xl,
                           const float* __restrict__ xr,
                           const float* __restrict__ att,
                           const int* __restrict__ ei,
                           int H, int C,
                           float* __restrict__ escore,
                           unsigned* __restrict__ mmax) {
    int wid  = (blockIdx.x * blockDim.x + threadIdx.x) >> 5;
    int lane = threadIdx.x & 31;
    if (wid >= EP_ * H) return;
    int e = wid / H, h = wid - e * H;
    int src, dst;
    if (e < EE) { src = ei[e]; dst = ei[EE + e]; }
    else        { src = dst = e - EE; }
    const int HC = H * C;
    const float* pl = xl + (size_t)src * HC + h * C;
    const float* pr = xr + (size_t)dst * HC + h * C;
    const float* pa = att + h * C;
    float s = 0.f;
    for (int c = lane; c < C; c += 32) {
        float v = pl[c] + pr[c];
        v = (v > 0.f) ? v : 0.2f * v;
        s += v * pa[c];
    }
#pragma unroll
    for (int off = 16; off > 0; off >>= 1) s += __shfl_xor(s, off, 32);
    if (lane == 0) {
        escore[wid] = s;
        atomicMax(mmax + (size_t)dst * H + h, encf(s));
    }
}

__global__ void gat_expsum(float* __restrict__ escore,
                           const unsigned* __restrict__ mmax,
                           float* __restrict__ ssum,
                           const int* __restrict__ ei, int H) {
    int gid = blockIdx.x * blockDim.x + threadIdx.x;
    if (gid >= EP_ * H) return;
    int e = gid / H, h = gid - e * H;
    int dst = (e < EE) ? ei[EE + e] : (e - EE);
    float m = decf(mmax[(size_t)dst * H + h]);
    float p = __expf(escore[gid] - m);
    escore[gid] = p;
    atomicAdd(ssum + (size_t)dst * H + h, p);
}

__global__ void gat_scatter(const float* __restrict__ xl,
                            const float* __restrict__ escore,
                            const float* __restrict__ ssum,
                            const int* __restrict__ ei,
                            int H, int C,
                            float* __restrict__ out) {
    long long gid = (long long)blockIdx.x * blockDim.x + threadIdx.x;
    const int HC = H * C;
    if (gid >= (long long)EP_ * HC) return;
    int e = (int)(gid / HC);
    int j = (int)(gid - (long long)e * HC);
    int h = j / C;
    int src, dst;
    if (e < EE) { src = ei[e]; dst = ei[EE + e]; }
    else        { src = dst = e - EE; }
    float alpha = escore[(size_t)e * H + h] / ssum[(size_t)dst * H + h];
    atomicAdd(out + (size_t)dst * HC + j, xl[(size_t)src * HC + j] * alpha);
}

// ---------------------------------------------------------------------------
// Launch
// ---------------------------------------------------------------------------
static inline int cdiv(long long a, long long b) { return (int)((a + b - 1) / b); }

extern "C" void kernel_launch(void* const* d_in, const int* in_sizes, int n_in,
                              void* d_out, int out_size, void* d_ws, size_t ws_size,
                              hipStream_t stream) {
    const float* x     = (const float*)d_in[0];
    const int*   ei    = (const int*)  d_in[1];
    const float* w1_l  = (const float*)d_in[2];
    const float* b1_l  = (const float*)d_in[3];
    const float* w1_r  = (const float*)d_in[4];
    const float* b1_r  = (const float*)d_in[5];
    const float* att1  = (const float*)d_in[6];
    const float* bias1 = (const float*)d_in[7];
    const float* w2_l  = (const float*)d_in[8];
    const float* b2_l  = (const float*)d_in[9];
    const float* w2_r  = (const float*)d_in[10];
    const float* b2_r  = (const float*)d_in[11];
    const float* att2  = (const float*)d_in[12];
    const float* bias2 = (const float*)d_in[13];
    const float* w_lin = (const float*)d_in[14];
    const float* b_lin = (const float*)d_in[15];
    float* out = (float*)d_out;

    // ---- workspace layout (bytes) ----
    char* ws = (char*)d_ws;
    size_t off = 0;
    auto take = [&](size_t bytes) { char* p = ws + off; off += (bytes + 255) & ~(size_t)255; return p; };
    unsigned short* X16  = (unsigned short*)take((size_t)NN * DD * 2);   // x bf16; reused as h2 bf16
    unsigned short* WT   = (unsigned short*)take((size_t)F1_ * DD * 2);  // 512x256 bf16 (max weight)
    float*          XL   = (float*)take((size_t)NN * F1_ * 4);
    float*          XR   = (float*)take((size_t)NN * F1_ * 4);
    float*          H1   = (float*)take((size_t)NN * F1_ * 4);
    unsigned short* H116 = (unsigned short*)take((size_t)NN * F1_ * 2);
    float*          H2   = (float*)take((size_t)NN * OUT_ * 4);
    float*          EBUF = (float*)take((size_t)EP_ * HEADS_ * 4);
    unsigned*       MBUF = (unsigned*)take((size_t)NN * HEADS_ * 4);
    float*          SBUF = (float*)take((size_t)NN * HEADS_ * 4);
    unsigned short* H216 = X16;  // reuse: x bf16 dead after layer-1 GEMMs
    (void)ws_size; (void)in_sizes; (void)n_in; (void)out_size;

    const unsigned ENC_NEG_INF = 0x007fffffu;  // encf(-inf)
    const dim3 blk(256);

    // x -> bf16
    convert_bf16<<<cdiv((long long)NN * DD, 256), blk, 0, stream>>>(x, X16, NN * DD);
    // h1 accumulator = 0
    fill_u32<<<cdiv((long long)NN * F1_, 256), blk, 0, stream>>>((unsigned*)H1, 0u, NN * F1_);

    // ================= layer 1 (H=8, C=64) =================
    for (int t = 0; t < TT; ++t) {
        const int* eit = ei + (size_t)t * 2 * EE;
        // xl = x @ w1_l[t] + b1_l[t]
        convert_transpose_bf16<<<cdiv((long long)DD * F1_, 256), blk, 0, stream>>>(
            w1_l + (size_t)t * DD * F1_, WT, DD, F1_);
        wmma_gemm_bias<<<dim3(cdiv(NN, 128), F1_ / 64), blk, 0, stream>>>(
            X16, WT, b1_l + (size_t)t * F1_, XL, NN, F1_, DD);
        // xr = x @ w1_r[t] + b1_r[t]
        convert_transpose_bf16<<<cdiv((long long)DD * F1_, 256), blk, 0, stream>>>(
            w1_r + (size_t)t * DD * F1_, WT, DD, F1_);
        wmma_gemm_bias<<<dim3(cdiv(NN, 128), F1_ / 64), blk, 0, stream>>>(
            X16, WT, b1_r + (size_t)t * F1_, XR, NN, F1_, DD);

        fill_u32<<<cdiv((long long)NN * HEADS_, 256), blk, 0, stream>>>(MBUF, ENC_NEG_INF, NN * HEADS_);
        fill_u32<<<cdiv((long long)NN * HEADS_, 256), blk, 0, stream>>>((unsigned*)SBUF, 0u, NN * HEADS_);

        gat_scores<<<cdiv((long long)EP_ * HEADS_ * 32, 256), blk, 0, stream>>>(
            XL, XR, att1 + (size_t)t * HEADS_ * HID_, eit, HEADS_, HID_, EBUF, MBUF);
        gat_expsum<<<cdiv((long long)EP_ * HEADS_, 256), blk, 0, stream>>>(
            EBUF, MBUF, SBUF, eit, HEADS_);
        gat_scatter<<<cdiv((long long)EP_ * F1_, 256), blk, 0, stream>>>(
            XL, EBUF, SBUF, eit, HEADS_, HID_, H1);
    }
    // h1 = elu(h1 + bias1[0] + bias1[1]) -> bf16
    elu_bias_convert<<<cdiv((long long)NN * F1_, 256), blk, 0, stream>>>(
        H1, bias1, bias1 + F1_, H116, NN * F1_, F1_);

    // h2 accumulator = 0
    fill_u32<<<cdiv((long long)NN * OUT_, 256), blk, 0, stream>>>((unsigned*)H2, 0u, NN * OUT_);

    // ================= layer 2 (H=1, C=256) =================
    for (int t = 0; t < TT; ++t) {
        const int* eit = ei + (size_t)t * 2 * EE;
        convert_transpose_bf16<<<cdiv((long long)F1_ * OUT_, 256), blk, 0, stream>>>(
            w2_l + (size_t)t * F1_ * OUT_, WT, F1_, OUT_);
        wmma_gemm_bias<<<dim3(cdiv(NN, 128), OUT_ / 64), blk, 0, stream>>>(
            H116, WT, b2_l + (size_t)t * OUT_, XL, NN, OUT_, F1_);
        convert_transpose_bf16<<<cdiv((long long)F1_ * OUT_, 256), blk, 0, stream>>>(
            w2_r + (size_t)t * F1_ * OUT_, WT, F1_, OUT_);
        wmma_gemm_bias<<<dim3(cdiv(NN, 128), OUT_ / 64), blk, 0, stream>>>(
            H116, WT, b2_r + (size_t)t * OUT_, XR, NN, OUT_, F1_);

        fill_u32<<<cdiv(NN, 256), blk, 0, stream>>>(MBUF, ENC_NEG_INF, NN);
        fill_u32<<<cdiv(NN, 256), blk, 0, stream>>>((unsigned*)SBUF, 0u, NN);

        gat_scores<<<cdiv((long long)EP_ * 32, 256), blk, 0, stream>>>(
            XL, XR, att2 + (size_t)t * OUT_, eit, 1, OUT_, EBUF, MBUF);
        gat_expsum<<<cdiv(EP_, 256), blk, 0, stream>>>(
            EBUF, MBUF, SBUF, eit, 1);
        gat_scatter<<<cdiv((long long)EP_ * OUT_, 256), blk, 0, stream>>>(
            XL, EBUF, SBUF, eit, 1, OUT_, H2);
    }
    // h2 = elu(h2 + bias2[0] + bias2[1]) -> bf16 (into reused X16 region)
    elu_bias_convert<<<cdiv((long long)NN * OUT_, 256), blk, 0, stream>>>(
        H2, bias2, bias2 + OUT_, H216, NN * OUT_, OUT_);

    // out = h2 @ w_lin + b_lin
    convert_transpose_bf16<<<cdiv((long long)OUT_ * OUT_, 256), blk, 0, stream>>>(
        w_lin, WT, OUT_, OUT_);
    wmma_gemm_bias<<<dim3(cdiv(NN, 128), OUT_ / 64), blk, 0, stream>>>(
        H216, WT, b_lin, out, NN, OUT_, OUT_);
}
